// ContentPopularityJointAttention_61521111548329
// MI455X (gfx1250) — matrix operand order repaired and checked by reading, not compiled, not run on hardware
//
#include <hip/hip_runtime.h>

typedef __attribute__((ext_vector_type(16))) __bf16 v16bf;
typedef __attribute__((ext_vector_type(8)))  __bf16 v8bf;
typedef __attribute__((ext_vector_type(8)))  float  v8f;

#define N_ART   50
#define M_SIZE  256
#define LDB     264   // LDS row stride in bf16 elems: 256 + 8 skew (4 dwords) -> conflict-free b128 LDS reads

// CDNA5 has a hardware tanh transcendental (v_tanh_f32); use it when the builtin exists.
static __device__ __forceinline__ float hw_tanhf(float x) {
#if defined(__has_builtin)
#if __has_builtin(__builtin_amdgcn_tanhf)
    return __builtin_amdgcn_tanhf(x);
#else
    return tanhf(x);
#endif
#else
    return tanhf(x);
#endif
}

// Prep: Wu (512x512 f32, [d][k] row-major) -> WuT (512x512 bf16, [k][d] row-major)
__global__ void wu_transpose_bf16(const float* __restrict__ Wu, __bf16* __restrict__ WuT) {
    int idx = blockIdx.x * blockDim.x + threadIdx.x;  // 512*512 threads
    int k = idx >> 9;
    int d = idx & 511;
    WuT[k * 512 + d] = (__bf16)Wu[d * 512 + k];
}

__global__ __launch_bounds__(256) void cpja_kernel(
    const float* __restrict__ m, const float* __restrict__ p,
    const __bf16* __restrict__ WuT, const float* __restrict__ bvec,
    float* __restrict__ out) {
    __shared__ __bf16 s_mp[64 * LDB];   // one K=256 slab of mp, bf16, rows padded to 64
    __shared__ float  s_scores[64];

    const int tid   = threadIdx.x;
    const int b     = blockIdx.x;
    const int wid   = tid >> 5;          // 8 waves
    const int lane  = tid & 31;
    const int half  = lane >> 4;
    const int mrow  = lane & 15;
    const int colBase = wid * 64;        // each wave: 4 output-column tiles

    if (tid < 64) s_scores[tid] = 0.0f;

    v8f acc[4][4] = {};                  // 4 M-tiles x 4 N-tiles, f32 accum (128 VGPRs)

    const float* srcs[2] = { m + (size_t)b * N_ART * M_SIZE,
                             p + (size_t)b * N_ART * M_SIZE };

    #pragma unroll
    for (int phase = 0; phase < 2; ++phase) {
        if (phase) __syncthreads();      // all waves done with previous slab
        const float* src = srcs[phase];
        // fill LDS slab: 64 rows x 256 cols (rows >= 50 zeroed); 4 f32 -> 4 bf16 per thread/iter
        #pragma unroll
        for (int it = 0; it < 16; ++it) {
            int i4  = (it * 256 + tid) * 4;
            int row = i4 >> 8;
            int col = i4 & 255;
            float4 v = {0.f, 0.f, 0.f, 0.f};
            if (row < N_ART) v = *(const float4*)(src + row * 256 + col);
            __bf16* dst = &s_mp[row * LDB + col];
            dst[0] = (__bf16)v.x; dst[1] = (__bf16)v.y;
            dst[2] = (__bf16)v.z; dst[3] = (__bf16)v.w;
        }
        __syncthreads();

        const int k0g = phase * 256;     // global K offset of this slab
        #pragma unroll
        for (int kt = 0; kt < 8; ++kt) {
            const int kl = kt * 32;
            // A fragments (16x32 bf16 layout): lanes 0-15 K{0..15}, lanes 16-31 K{16..31}
            v16bf afrag[4];
            #pragma unroll
            for (int mt = 0; mt < 4; ++mt) {
                const __bf16* rp = &s_mp[(mt * 16 + mrow) * LDB + kl + half * 16];
                v8bf lo = *(const v8bf*)(rp);        // K base..base+7
                v8bf hi = *(const v8bf*)(rp + 8);    // K base+8..base+15
                afrag[mt] = __builtin_shufflevector(lo, hi,
                    0,1,2,3,4,5,6,7,8,9,10,11,12,13,14,15);
            }
            #pragma unroll
            for (int ct = 0; ct < 4; ++ct) {
                // B fragment (32x16): lane = output column, 16 contiguous K per lane (half selects K 0-15 / 16-31)
                const __bf16* bp = &WuT[(size_t)(colBase + ct * 16 + mrow) * 512
                                        + k0g + kl + half * 16];
                v16bf bfrag = *(const v16bf*)(bp);
                #pragma unroll
                for (int mt = 0; mt < 4; ++mt) {
                    acc[mt][ct] = __builtin_amdgcn_wmma_f32_16x16x32_bf16(
                        false, afrag[mt], false, bfrag,
                        (short)0, acc[mt][ct], false, false);
                }
            }
        }
    }

    // scores: s[M] += sum_N tanh(h[M][N]) * b[N]   (f32, hardware v_tanh_f32)
    float bv[4];
    #pragma unroll
    for (int ct = 0; ct < 4; ++ct) bv[ct] = bvec[colBase + ct * 16 + mrow];

    #pragma unroll
    for (int mt = 0; mt < 4; ++mt) {
        #pragma unroll
        for (int j = 0; j < 8; ++j) {    // C/D layout: row M = mt*16 + half*8 + j, col N = mrow
            float t = 0.0f;
            #pragma unroll
            for (int ct = 0; ct < 4; ++ct) t += hw_tanhf(acc[mt][ct][j]) * bv[ct];
            t += __shfl_xor(t, 1, 32);
            t += __shfl_xor(t, 2, 32);
            t += __shfl_xor(t, 4, 32);
            t += __shfl_xor(t, 8, 32);   // reduced across the 16 N-columns of the half-wave
            if (mrow == 0) atomicAdd(&s_scores[mt * 16 + half * 8 + j], t);
        }
    }
    __syncthreads();

    float total = 0.0f;
    #pragma unroll
    for (int i = 0; i < 64; ++i) total += s_scores[i];   // padded rows contribute exactly 0
    const float inv = 1.0f / total;

    // pooling: u[d] = inv * sum_n s[n] * m[b][n][d]  (m rows are L2-hot from the slab load)
    const float* mp0 = m + (size_t)b * N_ART * M_SIZE + tid;
    float u = 0.0f;
    #pragma unroll 5
    for (int n = 0; n < N_ART; ++n) u += s_scores[n] * mp0[n * M_SIZE];
    out[(size_t)b * M_SIZE + tid] = u * inv;
}

extern "C" void kernel_launch(void* const* d_in, const int* in_sizes, int n_in,
                              void* d_out, int out_size, void* d_ws, size_t ws_size,
                              hipStream_t stream) {
    const float* m  = (const float*)d_in[0];   // (4096, 50, 256) f32
    const float* p  = (const float*)d_in[1];   // (4096, 50, 256) f32
    const float* Wu = (const float*)d_in[2];   // (512, 512) f32
    const float* bb = (const float*)d_in[3];   // (512,) f32
    __bf16* WuT = (__bf16*)d_ws;               // 512 KB scratch

    wu_transpose_bf16<<<(512 * 512) / 256, 256, 0, stream>>>(Wu, WuT);
    cpja_kernel<<<4096, 256, 0, stream>>>(m, p, WuT, bb, (float*)d_out);
}
